// Retention_73083163508859
// MI455X (gfx1250) — matrix operand order repaired and checked by reading, not compiled, not run on hardware
//
#include <hip/hip_runtime.h>
#include <hip/hip_bf16.h>
#include <math.h>

#define BB 8
#define SS 2048
#define DD 256

typedef __attribute__((ext_vector_type(16))) __bf16 v16bf;
typedef __attribute__((ext_vector_type(8)))  float  v8f;
typedef __attribute__((ext_vector_type(2)))  __bf16 v2bf;

union FragBF { v16bf v; unsigned u[8]; };

__device__ __forceinline__ unsigned pack2(float lo, float hi) {
  v2bf p = {(__bf16)lo, (__bf16)hi};            // -> v_cvt_pk_bf16_f32
  return __builtin_bit_cast(unsigned, p);
}
__device__ __forceinline__ unsigned short bf16bits(float x) {
  return __builtin_bit_cast(unsigned short, (__bf16)x);
}
// 8 contiguous dwords -> fragment (contiguous k = 16*lh + s order: base already includes 8*lh)
__device__ __forceinline__ void ld8(FragBF& f, const unsigned* base) {
  uint4 a = *(const uint4*)(base);
  uint4 b = *(const uint4*)(base + 4);
  f.u[0] = a.x; f.u[1] = a.y; f.u[2] = a.z; f.u[3] = a.w;
  f.u[4] = b.x; f.u[5] = b.y; f.u[6] = b.z; f.u[7] = b.w;
}
// A-operand layout: two contiguous 4-dword groups at +4*lh and +8+4*lh
__device__ __forceinline__ void ldA(FragBF& f, const unsigned* base, int lh) {
  uint4 a = *(const uint4*)(base + 4 * lh);
  uint4 b = *(const uint4*)(base + 8 + 4 * lh);
  f.u[0] = a.x; f.u[1] = a.y; f.u[2] = a.z; f.u[3] = a.w;
  f.u[4] = b.x; f.u[5] = b.y; f.u[6] = b.z; f.u[7] = b.w;
}

// ---------------------------------------------------------------------------
// Prep: Wt[z][n][k] = bf16(W[z][k][n])  (B-fragment-friendly layout)
// ---------------------------------------------------------------------------
__global__ __launch_bounds__(256) void wprep_kernel(
    const float* __restrict__ Wq, const float* __restrict__ Wk,
    const float* __restrict__ Wv, unsigned short* __restrict__ Wt)
{
  const int z = blockIdx.z;
  const float* W = (z == 0) ? Wq : (z == 1) ? Wk : Wv;
  const int idx = blockIdx.x * 256 + threadIdx.x;   // 0..65535
  const int k = idx >> 8, n = idx & 255;            // coalesced read of W
  Wt[(size_t)z * DD * DD + (size_t)n * DD + k] = bf16bits(W[(size_t)k * DD + n]);
}

// ---------------------------------------------------------------------------
// Projection: C[16384,256] = X @ W -> bf16.  z=0: Q row-major, z=1: K
// row-major, z=2: V transposed Vt[b][d][s].  One wave = 16x64 tile.
// ---------------------------------------------------------------------------
__global__ __launch_bounds__(256) void proj_kernel(
    const float* __restrict__ xq, const float* __restrict__ xk, const float* __restrict__ xv,
    const unsigned short* __restrict__ Wt,
    unsigned short* __restrict__ Qb, unsigned short* __restrict__ Kb,
    unsigned short* __restrict__ Vt)
{
  const int lane = threadIdx.x & 31;
  const int w    = threadIdx.x >> 5;
  const int z    = blockIdx.z;
  const float* X = (z == 0) ? xq : (z == 1) ? xk : xv;
  const unsigned short* Wz = Wt + (size_t)z * DD * DD;

  const int t   = blockIdx.x * 8 + w;  // 0..4095
  const int i0  = (t >> 2) * 16;
  const int n0b = (t & 3) * 64;
  const int lm = lane & 15, lh = lane >> 4;

  v8f acc[4] = {};
  for (int kc = 0; kc < 8; ++kc) {
    // A fragment from f32 X: 4x float4 + packed cvt
    FragBF a;
    const float4* xr = (const float4*)(X + (size_t)(i0 + lm) * DD + kc * 32);
    float4 f0 = xr[2 * lh];
    float4 f1 = xr[2 * lh + 1];
    float4 f2 = xr[4 + 2 * lh];
    float4 f3 = xr[5 + 2 * lh];
    a.u[0] = pack2(f0.x, f0.y); a.u[1] = pack2(f0.z, f0.w);
    a.u[2] = pack2(f1.x, f1.y); a.u[3] = pack2(f1.z, f1.w);
    a.u[4] = pack2(f2.x, f2.y); a.u[5] = pack2(f2.z, f2.w);
    a.u[6] = pack2(f3.x, f3.y); a.u[7] = pack2(f3.z, f3.w);
#pragma unroll
    for (int nt = 0; nt < 4; ++nt) {
      FragBF bf;
      const unsigned* wrow = (const unsigned*)(Wz + (size_t)(n0b + nt * 16 + lm) * DD);
      ldA(bf, wrow + kc * 16, lh);
      acc[nt] = __builtin_amdgcn_wmma_f32_16x16x32_bf16(
          false, a.v, false, bf.v, (short)0, acc[nt], false, false);
    }
  }
  // C layout: vgpr r, lane l -> (M = r + 8*lh, N = lm).
  // z is block-uniform: hoist the destination select out of the store loops.
  if (z == 0) {
    unsigned short* qp = Qb + (size_t)(i0 + 8 * lh) * DD + lm;
#pragma unroll
    for (int nt = 0; nt < 4; ++nt)
#pragma unroll
      for (int r = 0; r < 8; ++r)
        qp[(size_t)r * DD + n0b + nt * 16] = bf16bits(acc[nt][r]);
  } else if (z == 1) {
    unsigned short* kp = Kb + (size_t)(i0 + 8 * lh) * DD + lm;
#pragma unroll
    for (int nt = 0; nt < 4; ++nt)
#pragma unroll
      for (int r = 0; r < 8; ++r)
        kp[(size_t)r * DD + n0b + nt * 16] = bf16bits(acc[nt][r]);
  } else {
    const int b = i0 >> 11, s0 = (i0 & (SS - 1)) + 8 * lh;   // 16-row tile never splits b
    unsigned short* vp = Vt + ((size_t)b * DD + n0b + lm) * SS + s0;
#pragma unroll
    for (int nt = 0; nt < 4; ++nt)
#pragma unroll
      for (int r = 0; r < 8; ++r)
        vp[(size_t)(nt * 16) * SS + r] = bf16bits(acc[nt][r]);
  }
}

// ---------------------------------------------------------------------------
// Pass A: rowsum[b][i] = sum_j scores[b][i][j]  (post decay+scale, causal)
// ---------------------------------------------------------------------------
__global__ __launch_bounds__(256) void rowsum_kernel(
    const unsigned short* __restrict__ Qb, const unsigned short* __restrict__ Kb,
    float* __restrict__ rowsum, float scale, float lg2g)
{
  const int lane = threadIdx.x & 31;
  const int w    = threadIdx.x >> 5;
  const int g = blockIdx.x * 8 + w;          // 0..1023
  const int b = g >> 7, it = g & 127;
  const int i0 = it * 16;
  const int lm = lane & 15, lh = lane >> 4;

  float gr[8];
  gr[0] = 1.f;
#pragma unroll
  for (int r = 1; r < 8; ++r) gr[r] = gr[r - 1] * 0.9865f;

  const unsigned* Qrow = (const unsigned*)(Qb + ((size_t)b * SS + i0 + lm) * DD);
  FragBF qa[8];
#pragma unroll
  for (int kc = 0; kc < 8; ++kc) ldA(qa[kc], Qrow + kc * 16, lh);

  float rs[8] = {0.f, 0.f, 0.f, 0.f, 0.f, 0.f, 0.f, 0.f};
  // sub-diagonal tiles: no mask needed
  for (int jt = 0; jt < it; ++jt) {
    const int j0 = jt * 16;
    const unsigned* Krow = (const unsigned*)(Kb + ((size_t)b * SS + j0 + lm) * DD);
    __builtin_prefetch((const void*)(Kb + ((size_t)b * SS + j0 + 16 + lm) * DD), 0, 1);
    v8f c = {};
#pragma unroll
    for (int kc = 0; kc < 8; ++kc) {
      FragBF kb;
      ldA(kb, Krow + kc * 16, lh);
      c = __builtin_amdgcn_wmma_f32_16x16x32_bf16(
          false, qa[kc].v, false, kb.v, (short)0, c, false, false);
    }
    const float base = scale * exp2f(lg2g * (float)(i0 - j0 + 8 * lh - lm));
#pragma unroll
    for (int r = 0; r < 8; ++r) rs[r] += c[r] * base * gr[r];
  }
  // diagonal tile: masked
  {
    const unsigned* Krow = (const unsigned*)(Kb + ((size_t)b * SS + i0 + lm) * DD);
    v8f c = {};
#pragma unroll
    for (int kc = 0; kc < 8; ++kc) {
      FragBF kb;
      ldA(kb, Krow + kc * 16, lh);
      c = __builtin_amdgcn_wmma_f32_16x16x32_bf16(
          false, qa[kc].v, false, kb.v, (short)0, c, false, false);
    }
    const float base = scale * exp2f(lg2g * (float)(8 * lh - lm));
#pragma unroll
    for (int r = 0; r < 8; ++r)
      rs[r] += (r + 8 * lh >= lm) ? c[r] * base * gr[r] : 0.f;
  }
  // reduce over the 16 columns held by each 16-lane half
#pragma unroll
  for (int r = 0; r < 8; ++r) {
    float v = rs[r];
    v += __shfl_xor(v, 1, 32);
    v += __shfl_xor(v, 2, 32);
    v += __shfl_xor(v, 4, 32);
    v += __shfl_xor(v, 8, 32);
    rs[r] = v;
  }
  if (lm == 0) {
#pragma unroll
    for (int r = 0; r < 8; ++r)
      rowsum[(size_t)b * SS + i0 + r + 8 * lh] = rs[r];
  }
}

// ---------------------------------------------------------------------------
// Pass B: out = P @ V, P[i][j] = scores[i][j] / max(|rowsum[b][j]|, 1)
// ---------------------------------------------------------------------------
__global__ __launch_bounds__(256) void out_kernel(
    const unsigned short* __restrict__ Qb, const unsigned short* __restrict__ Kb,
    const unsigned short* __restrict__ Vt, const float* __restrict__ rowsum,
    float* __restrict__ out, float scale, float lg2g)
{
  __shared__ __align__(16) unsigned short plds[8][16][32];  // per-wave P tile
  const int lane = threadIdx.x & 31;
  const int w    = threadIdx.x >> 5;
  const int g = blockIdx.x * 8 + w;
  const int b = g >> 7, it = g & 127;
  const int i0 = it * 16;
  const int lm = lane & 15, lh = lane >> 4;

  float gr[8];
  gr[0] = 1.f;
#pragma unroll
  for (int r = 1; r < 8; ++r) gr[r] = gr[r - 1] * 0.9865f;

  const unsigned* Qrow = (const unsigned*)(Qb + ((size_t)b * SS + i0 + lm) * DD);
  FragBF qa[8];
#pragma unroll
  for (int kc = 0; kc < 8; ++kc) ldA(qa[kc], Qrow + kc * 16, lh);

  v8f acc[16] = {};
  const int nch = (i0 + 47) >> 5;     // #32-wide j-chunks covering j <= i0+15
  for (int jc = 0; jc < nch; ++jc) {
    const int j0c = jc * 32;
    if (jc + 1 < nch)
      __builtin_prefetch((const void*)(Kb + ((size_t)b * SS + j0c + 32 + lm) * DD), 0, 1);
#pragma unroll
    for (int t = 0; t < 2; ++t) {
      const int j0 = j0c + t * 16;
      if (j0 <= i0 + 15) {
        const unsigned* Krow = (const unsigned*)(Kb + ((size_t)b * SS + j0 + lm) * DD);
        v8f c = {};
#pragma unroll
        for (int kc = 0; kc < 8; ++kc) {
          FragBF kb;
          ldA(kb, Krow + kc * 16, lh);
          c = __builtin_amdgcn_wmma_f32_16x16x32_bf16(
              false, qa[kc].v, false, kb.v, (short)0, c, false, false);
        }
        const float rden = 1.f / fmaxf(fabsf(rowsum[(size_t)b * SS + j0 + lm]), 1.0f);
        const float base = rden * scale * exp2f(lg2g * (float)(i0 - j0 + 8 * lh - lm));
        if (j0 < i0) {          // fully sub-diagonal: no mask
#pragma unroll
          for (int r = 0; r < 8; ++r)
            plds[w][r + 8 * lh][t * 16 + lm] = bf16bits(c[r] * base * gr[r]);
        } else {                // diagonal tile: mask j > i
#pragma unroll
          for (int r = 0; r < 8; ++r) {
            const float p = (r + 8 * lh >= lm) ? c[r] * base * gr[r] : 0.f;
            plds[w][r + 8 * lh][t * 16 + lm] = bf16bits(p);
          }
        }
      } else {
#pragma unroll
        for (int r = 0; r < 8; ++r)
          plds[w][r + 8 * lh][t * 16 + lm] = 0;
      }
    }
    // A-layout fragment of P from LDS (same-wave LDS ops are in-order)
    FragBF pa;
    ldA(pa, (const unsigned*)plds[w][lm], lh);
    // P(16x32) @ V(32x256): Vt is [b][d][s], two 4-dword groups per frag half (ldA order)
#pragma unroll
    for (int nt = 0; nt < 16; ++nt) {
      const unsigned* Vrow =
          (const unsigned*)(Vt + ((size_t)b * DD + nt * 16 + lm) * SS + j0c);
      FragBF vb;
      ldA(vb, Vrow, lh);
      acc[nt] = __builtin_amdgcn_wmma_f32_16x16x32_bf16(
          false, pa.v, false, vb.v, (short)0, acc[nt], false, false);
    }
  }
  float* op = out + ((size_t)b * SS + i0 + 8 * lh) * DD + lm;
#pragma unroll
  for (int nt = 0; nt < 16; ++nt)
#pragma unroll
    for (int r = 0; r < 8; ++r)
      op[(size_t)r * DD + nt * 16] = acc[nt][r];
}

extern "C" void kernel_launch(void* const* d_in, const int* in_sizes, int n_in,
                              void* d_out, int out_size, void* d_ws, size_t ws_size,
                              hipStream_t stream) {
  const float* xq = (const float*)d_in[0];
  const float* xk = (const float*)d_in[1];
  const float* xv = (const float*)d_in[2];
  const float* Wq = (const float*)d_in[3];
  const float* Wk = (const float*)d_in[4];
  const float* Wv = (const float*)d_in[5];
  float* out = (float*)d_out;

  char* ws = (char*)d_ws;
  const size_t szQ = (size_t)BB * SS * DD * sizeof(unsigned short);  // 8.39 MB
  unsigned short* Qb = (unsigned short*)ws;
  unsigned short* Kb = (unsigned short*)(ws + szQ);
  unsigned short* Vt = (unsigned short*)(ws + 2 * szQ);
  float* rsum        = (float*)(ws + 3 * szQ);
  unsigned short* Wt = (unsigned short*)(ws + 3 * szQ + (size_t)BB * SS * sizeof(float));

  // Host-side deterministic scalars: D normalizer and log2(gamma)
  double sum = 0.0, p = 1.0;
  const double gamma = 0.9865;
  for (int d = 0; d < SS; ++d) { sum += (double)(SS - d) * p; p *= gamma; }
  const float scale = (float)((1.0 / 16.0) / sqrt(sum));  // 1/sqrt(DIM) * 1/sqrt(sum|D|)
  const float lg2g  = (float)(log(gamma) / log(2.0));

  dim3 wgrid(256, 1, 3);
  wprep_kernel<<<wgrid, 256, 0, stream>>>(Wq, Wk, Wv, Wt);
  dim3 pgrid(512, 1, 3);
  proj_kernel<<<pgrid, 256, 0, stream>>>(xq, xk, xv, Wt, Qb, Kb, Vt);
  rowsum_kernel<<<128, 256, 0, stream>>>(Qb, Kb, rsum, scale, lg2g);
  out_kernel<<<128, 256, 0, stream>>>(Qb, Kb, Vt, rsum, out, scale, lg2g);
}
